// linear_nce_32744830664773
// MI455X (gfx1250) — matrix operand order repaired and checked by reading, not compiled, not run on hardware
//
#include <hip/hip_runtime.h>

typedef unsigned short u16;
typedef __attribute__((ext_vector_type(16))) __bf16 v16bf;
typedef __attribute__((ext_vector_type(8)))  float  v8f;
typedef __attribute__((ext_vector_type(4)))  unsigned v4u;
typedef __attribute__((ext_vector_type(8)))  unsigned v8u;

constexpr int NN = 8192;   // rows (batch)
constexpr int HH = 1024;   // hidden
constexpr int KK = 4096;   // noise samples

#define BM 128
#define BN 128
#define BK 32

struct TwoU4 { uint4 x, y; };
static __device__ __forceinline__ v16bf make_frag(uint4 x, uint4 y) {
    TwoU4 s{x, y};
    return __builtin_bit_cast(v16bf, s);
}

static __device__ __forceinline__ u16 f2bf(float f) {
    unsigned u = __builtin_bit_cast(unsigned, f);
    unsigned r = u + 0x7fffu + ((u >> 16) & 1u);   // round-to-nearest-even
    return (u16)(r >> 16);
}

// ---- Tensor Data Mover descriptors (CDNA5 ISA) ----
// 2D tile: tile_rows x tile_d0 elements, element size 1<<dsz_code bytes,
// global row stride = stride0 elements.
static __device__ __forceinline__ void tdm_desc(v4u& g0, v8u& g1,
                                                unsigned lds_off,
                                                unsigned long long gaddr,
                                                unsigned dsz_code,
                                                unsigned tile_d0,
                                                unsigned tile_rows,
                                                unsigned tensor_d0,
                                                unsigned stride0) {
    g0.x = 1u;                                    // count=1 (valid user descriptor)
    g0.y = lds_off;                               // lds_addr (bytes)
    g0.z = (unsigned)gaddr;                       // global_addr[31:0]
    g0.w = (unsigned)(gaddr >> 32) | (2u << 30);  // global_addr[56:32] | type=2
    g1[0] = dsz_code << 16;                       // workgroup_mask=0, data_size
    g1[1] = (tensor_d0 & 0xFFFFu) << 16;          // tensor_dim0[15:0] @ bits 63:48
    g1[2] = (tensor_d0 >> 16) | (0x4000u << 16);  // dim0[31:16] | tensor_dim1[15:0]=16384
    g1[3] = (tile_d0 & 0xFFFFu) << 16;            // tensor_dim1[31:16]=0, tile_dim0
    g1[4] = tile_rows;                            // tile_dim1 = rows, tile_dim2 = 0
    g1[5] = stride0;                              // tensor_dim0_stride[31:0] (elements)
    g1[6] = 0u;
    g1[7] = 0u;
}

static __device__ __forceinline__ void tdm_load_2d(unsigned lds_off,
                                                   unsigned long long gaddr,
                                                   unsigned dsz_code, unsigned tile_d0,
                                                   unsigned tile_rows, unsigned tensor_d0,
                                                   unsigned stride0) {
    v4u g0; v8u g1;
    tdm_desc(g0, g1, lds_off, gaddr, dsz_code, tile_d0, tile_rows, tensor_d0, stride0);
    v4u gz = {0u, 0u, 0u, 0u};
    asm volatile("tensor_load_to_lds %0, %1, %2, %3"
                 :: "s"(g0), "s"(g1), "s"(gz), "s"(gz) : "memory");
}

static __device__ __forceinline__ void tdm_store_2d(unsigned lds_off,
                                                    unsigned long long gaddr,
                                                    unsigned dsz_code, unsigned tile_d0,
                                                    unsigned tile_rows, unsigned tensor_d0,
                                                    unsigned stride0) {
    v4u g0; v8u g1;
    tdm_desc(g0, g1, lds_off, gaddr, dsz_code, tile_d0, tile_rows, tensor_d0, stride0);
    v4u gz = {0u, 0u, 0u, 0u};
    asm volatile("tensor_store_from_lds %0, %1, %2, %3"
                 :: "s"(g0), "s"(g1), "s"(gz), "s"(gz) : "memory");
}

// ---------------- input fp32 -> bf16 ----------------
__global__ __launch_bounds__(256) void cvt_bf16(const float* __restrict__ src,
                                                u16* __restrict__ dst, int n) {
    int i = (blockIdx.x * blockDim.x + threadIdx.x) * 4;
    if (i >= n) return;
    float4 f = *(const float4*)(src + i);
    ushort4 o;
    o.x = f2bf(f.x); o.y = f2bf(f.y); o.z = f2bf(f.z); o.w = f2bf(f.w);
    *(ushort4*)(dst + i) = o;
}

// ---------------- gather weight[noise] -> bf16, plus b_noise / pn_noise ----------------
__global__ __launch_bounds__(256) void gather_noise(const float* __restrict__ weight,
                                                    const float* __restrict__ bias,
                                                    const float* __restrict__ up,
                                                    const int*   __restrict__ noise,
                                                    u16*  __restrict__ wns,
                                                    float* __restrict__ b_noise,
                                                    float* __restrict__ pn_noise) {
    int k = blockIdx.x;
    int r = noise[k];
    const float* src = weight + (size_t)r * HH;
    u16* dst = wns + (size_t)k * HH;
    int c = threadIdx.x * 4;                      // 256 threads * 4 = 1024 = HH
    float4 f = *(const float4*)(src + c);
    ushort4 o;
    o.x = f2bf(f.x); o.y = f2bf(f.y); o.z = f2bf(f.z); o.w = f2bf(f.w);
    *(ushort4*)(dst + c) = o;
    if (threadIdx.x == 0) {
        b_noise[k]  = bias[r];
        pn_noise[k] = up[r];
    }
}

// ---------------- pnn broadcast: pnn[n][k] = pn_noise[k], pure b128 stream ----------------
__global__ __launch_bounds__(256) void bcast_pnn(const float* __restrict__ pn_noise,
                                                 float* __restrict__ pnn) {
    size_t i = ((size_t)blockIdx.x * blockDim.x + threadIdx.x) * 4;
    int k = (int)(i & (size_t)(KK - 1));
    float4 v = *(const float4*)(pn_noise + k);
    *(float4*)(pnn + i) = v;
}

// ---------------- pmt / pnt : one wave32 per row ----------------
__global__ __launch_bounds__(256) void nce_target(const float* __restrict__ inp,
                                                  const float* __restrict__ weight,
                                                  const float* __restrict__ bias,
                                                  const float* __restrict__ up,
                                                  const int*   __restrict__ target,
                                                  float* __restrict__ pmt,
                                                  float* __restrict__ pnt) {
    int row  = (blockIdx.x * blockDim.x + threadIdx.x) >> 5;
    int lane = threadIdx.x & 31;
    if (row >= NN) return;
    int t = target[row];
    const float* x = inp    + (size_t)row * HH;
    const float* w = weight + (size_t)t   * HH;
    float s = 0.f;
    #pragma unroll
    for (int i = 0; i < HH; i += 128) {           // 32 lanes * float4 = 128 per step
        float4 a = *(const float4*)(x + i + lane * 4);
        float4 b = *(const float4*)(w + i + lane * 4);
        s += a.x * b.x + a.y * b.y + a.z * b.z + a.w * b.w;
    }
    #pragma unroll
    for (int off = 16; off; off >>= 1) s += __shfl_xor(s, off, 32);
    if (lane == 0) {
        pmt[row] = expf(s + bias[t]);
        pnt[row] = up[t];
    }
}

// ---------------- main GEMM: pmn = exp(A @ B^T + b) ----------------
// 128x128 block tile, 8 waves, each wave computes a 32x64 tile (2x4 WMMA accumulators).
// A/B K-chunks staged to double-buffered LDS by TENSOR_LOAD_TO_LDS (TENSORcnt);
// result tile staged to LDS and written back with one TENSOR_STORE_FROM_LDS DMA.
__global__ __launch_bounds__(256) void nce_gemm(const u16* __restrict__ Abf,   // [NN][HH]
                                                const u16* __restrict__ Bbf,   // [KK][HH]
                                                const float* __restrict__ b_noise,
                                                float* __restrict__ pmn) {
    __shared__ u16   As[2][BM * BK];   // 2 x 8 KB
    __shared__ u16   Bs[2][BN * BK];   // 2 x 8 KB
    __shared__ float Ct[BM * BN];      // 64 KB result tile

    const int tid  = threadIdx.x;
    const int lane = tid & 31;
    const int wave = tid >> 5;
    const int hi   = lane >> 4;        // half-wave select
    const int ln   = lane & 15;
    const int m0   = (wave & 3) * 32;  // wave tile origin in block
    const int n0   = (wave >> 2) * 64;
    const int blockM = blockIdx.y * BM;
    const int blockN = blockIdx.x * BN;

    v8f acc[2][4] = {};

    const unsigned ldsA0 = (unsigned)(unsigned long long)(void*)&As[0][0];
    const unsigned ldsA1 = (unsigned)(unsigned long long)(void*)&As[1][0];
    const unsigned ldsB0 = (unsigned)(unsigned long long)(void*)&Bs[0][0];
    const unsigned ldsB1 = (unsigned)(unsigned long long)(void*)&Bs[1][0];
    const unsigned ldsC  = (unsigned)(unsigned long long)(void*)&Ct[0];
    const unsigned long long baseA =
        (unsigned long long)(const void*)(Abf + (size_t)blockM * HH);
    const unsigned long long baseB =
        (unsigned long long)(const void*)(Bbf + (size_t)blockN * HH);

    if (wave == 0) {                       // prime buffer 0 with K-chunk 0
        tdm_load_2d(ldsA0, baseA, 1u, BK, BM, HH, HH);
        tdm_load_2d(ldsB0, baseB, 1u, BK, BN, HH, HH);
    }

    constexpr int NCHUNK = HH / BK;        // 32
    for (int chunk = 0; chunk < NCHUNK; ++chunk) {
        const int cur = chunk & 1;
        if (wave == 0) __builtin_amdgcn_s_wait_tensorcnt(0);
        __syncthreads();                   // publish TDM data; close WAR on other buffer
        if (wave == 0 && chunk + 1 < NCHUNK) {
            const unsigned long long koff = (unsigned long long)(chunk + 1) * BK * 2;
            tdm_load_2d(cur ? ldsA0 : ldsA1, baseA + koff, 1u, BK, BM, HH, HH);
            tdm_load_2d(cur ? ldsB0 : ldsB1, baseB + koff, 1u, BK, BN, HH, HH);
        }

        const u16* As_c = &As[cur][0];
        const u16* Bs_c = &Bs[cur][0];

        v16bf a[2], b[4];
        #pragma unroll
        for (int i = 0; i < 2; ++i) {
            // A lane(hi,m): K in [8*hi, 8*hi+8) and [16+8*hi, 24+8*hi)
            const u16* pa = As_c + (m0 + 16 * i + ln) * BK + hi * 8;
            a[i] = make_frag(*(const uint4*)pa, *(const uint4*)(pa + 16));
        }
        #pragma unroll
        for (int j = 0; j < 4; ++j) {
            // B lane(hi,n): K in [8*hi, 8*hi+8) and [16+8*hi, 24+8*hi)
            const u16* pb = Bs_c + (n0 + 16 * j + ln) * BK + hi * 8;
            b[j] = make_frag(*(const uint4*)pb, *(const uint4*)(pb + 16));
        }
        #pragma unroll
        for (int i = 0; i < 2; ++i)
            #pragma unroll
            for (int j = 0; j < 4; ++j)
                acc[i][j] = __builtin_amdgcn_wmma_f32_16x16x32_bf16(
                    false, a[i], false, b[j], (short)0, acc[i][j], false, false);
    }

    // epilogue: exp(acc + bias) staged into LDS tile, then one TDM store burst
    #pragma unroll
    for (int j = 0; j < 4; ++j) {
        const int cc = n0 + 16 * j + ln;           // column within block tile
        const float bn = b_noise[blockN + cc];
        #pragma unroll
        for (int i = 0; i < 2; ++i) {
            #pragma unroll
            for (int r = 0; r < 8; ++r) {
                const int rr = m0 + 16 * i + 8 * hi + r;   // row within block tile
                Ct[rr * BN + cc] = __expf(acc[i][j][r] + bn);
            }
        }
    }
    __syncthreads();                       // all lanes' DS stores visible before DMA
    if (wave == 0) {
        const unsigned long long gC =
            (unsigned long long)(const void*)(pmn + (size_t)blockM * KK + blockN);
        tdm_store_2d(ldsC, gC, 2u, BN, BM, KK, KK);
    }
    // S_ENDPGM performs an implicit wait-idle (covers outstanding TENSORcnt).
}

extern "C" void kernel_launch(void* const* d_in, const int* in_sizes, int n_in,
                              void* d_out, int out_size, void* d_ws, size_t ws_size,
                              hipStream_t stream) {
    const float* input  = (const float*)d_in[0];   // [NN,HH]
    const float* weight = (const float*)d_in[1];   // [V,HH]
    const float* bias   = (const float*)d_in[2];   // [V]
    const float* up     = (const float*)d_in[3];   // [V]
    const int*   target = (const int*)d_in[4];     // [NN]
    const int*   noise  = (const int*)d_in[5];     // [KK]

    float* out = (float*)d_out;
    float* pmt = out;                               // [NN]
    float* pnt = out + NN;                          // [NN]
    float* pmn = out + 2 * (size_t)NN;              // [NN,KK]
    float* pnn = pmn + (size_t)NN * KK;             // [NN,KK]

    u16*   in_bf    = (u16*)d_ws;                   // NN*HH bf16 = 16 MB
    u16*   wns      = in_bf + (size_t)NN * HH;      // KK*HH bf16 =  8 MB
    float* b_noise  = (float*)(wns + (size_t)KK * HH);
    float* pn_noise = b_noise + KK;

    cvt_bf16   <<<(NN * HH) / 1024, 256, 0, stream>>>(input, in_bf, NN * HH);
    gather_noise<<<KK, 256, 0, stream>>>(weight, bias, up, noise, wns, b_noise, pn_noise);
    nce_target <<<NN / 8, 256, 0, stream>>>(input, weight, bias, up, target, pmt, pnt);
    bcast_pnn  <<<(int)(((size_t)NN * KK) / 1024), 256, 0, stream>>>(pn_noise, pnn);

    dim3 grid(KK / BN, NN / BM);   // (32, 64)
    nce_gemm<<<grid, 256, 0, stream>>>(in_bf, wns, b_noise, pmn);
}
